// Listener_56736517980353
// MI455X (gfx1250) — compile-verified, matrix operand order checked by reading, not run on hardware
//
#include <hip/hip_runtime.h>
#include <hip/hip_bf16.h>
#include <cstdint>

typedef __attribute__((ext_vector_type(16))) _Float16 v16h;
typedef __attribute__((ext_vector_type(8)))  _Float16 v8h;
typedef __attribute__((ext_vector_type(8)))  float    v8f;
typedef int v4i __attribute__((__vector_size__(4 * sizeof(int))));

#define NB 32           // batch
#define TT 799          // time steps after convs
#define UU 512          // hidden units
#define GG 2048         // 4*U gates
#define DD0 608         // 19*32 conv feature dim
#define DY 1024         // 2*U
#define MMR (NB*TT)     // 25568 rows
#define KSTAGE 128      // K columns per async LDS stage in wgemm

#if __has_builtin(__builtin_amdgcn_global_load_async_to_lds_b128)
#define HAVE_ASYNC_LDS 1
#endif

// ---------------------------------------------------------------------------
// CDNA5 async global->LDS copy (ASYNCcnt path), with synchronous fallback
// ---------------------------------------------------------------------------
__device__ __forceinline__ void g2l_b128(const void* g, void* l) {
#if defined(HAVE_ASYNC_LDS)
  typedef __attribute__((address_space(1))) v4i* g128_t;
  typedef __attribute__((address_space(3))) v4i* l128_t;
  __builtin_amdgcn_global_load_async_to_lds_b128(
      (g128_t)(unsigned long long)(uintptr_t)g,
      (l128_t)(unsigned int)(uintptr_t)l, 0, 0);
#else
  *(v8h*)l = *(const v8h*)g;
#endif
}

template <int N>
__device__ __forceinline__ void wait_async() {
#if defined(HAVE_ASYNC_LDS)
#if __has_builtin(__builtin_amdgcn_s_wait_asynccnt)
  __builtin_amdgcn_s_wait_asynccnt(N);
#else
  asm volatile("s_wait_asynccnt %0" ::"i"(N) : "memory");
#endif
#endif
}

// ---------------------------------------------------------------------------
// WMMA helpers
// ---------------------------------------------------------------------------
__device__ __forceinline__ v8f wmma_f16(v16h a, v16h b, v8f c) {
  // (neg_a, A, neg_b, B, c_mod, C, reuse_a, reuse_b)
  return __builtin_amdgcn_wmma_f32_16x16x32_f16(false, a, false, b, (short)0, c,
                                                false, false);
}

// Load a 16x32 f16 fragment from a row-major matrix (row = lane%16).
// Per CDNA5 ISA: lane l (m = l&15, h = l>>4) holds
//   frag[i]   = M[m, k0 + h*8 + i],  frag[8+i] = M[m, k0 + 16 + h*8 + i]
// The same loader feeds B-fragments from a pre-transposed (N,K) weight matrix.
__device__ __forceinline__ v16h load_frag(const _Float16* base, int ld,
                                          int row0, int k0, int lane) {
  const _Float16* p =
      base + (size_t)(row0 + (lane & 15)) * ld + k0 + ((lane >> 4) << 3);
  v8h lo = *(const v8h*)p;
  v8h hi = *(const v8h*)(p + 16);
  return __builtin_shufflevector(lo, hi, 0, 1, 2, 3, 4, 5, 6, 7, 8, 9, 10, 11,
                                 12, 13, 14, 15);
}

// ---------------------------------------------------------------------------
// Mask: m[b,j] = any(audio[b, 4j..4j+4, 0..80) != 0)
// ---------------------------------------------------------------------------
__global__ void mask_kernel(const float* __restrict__ audio,
                            float* __restrict__ mw, float* __restrict__ mo) {
  int i = blockIdx.x * blockDim.x + threadIdx.x;
  if (i >= MMR) return;
  int b = i / TT, j = i % TT;
  const float* p = audio + ((size_t)b * 3200 + (size_t)j * 4) * 80;
  float any = 0.f;
  for (int q = 0; q < 320; ++q) {
    if (p[q] != 0.f) { any = 1.f; break; }
  }
  mw[i] = any;
  mo[i] = any;
}

// ---------------------------------------------------------------------------
// Convolutions (tiny compute; direct)
// ---------------------------------------------------------------------------
__global__ void conv1_kernel(const float* __restrict__ a,
                             const float* __restrict__ w,
                             const float* __restrict__ bi,
                             float* __restrict__ o) {
  long tid = (long)blockIdx.x * 256 + threadIdx.x;
  const long total = (long)NB * 1599 * 39 * 32;
  if (tid >= total) return;
  int co = (int)(tid & 31);
  int j = (int)((tid >> 5) % 39);
  int i = (int)((tid / (32 * 39)) % 1599);
  int b = (int)(tid / (32L * 39 * 1599));
  float acc = bi[co];
  for (int kh = 0; kh < 3; ++kh)
    for (int kw = 0; kw < 3; ++kw)
      acc += a[((size_t)b * 3200 + 2 * i + kh) * 80 + 2 * j + kw] *
             w[(kh * 3 + kw) * 32 + co];
  o[((size_t)b * 1599 + i) * 1248 + j * 32 + co] = acc;
}

__global__ void conv2_kernel(const float* __restrict__ x,
                             const float* __restrict__ w,
                             const float* __restrict__ bi,
                             _Float16* __restrict__ o) {
  long tid = (long)blockIdx.x * 256 + threadIdx.x;
  const long total = (long)NB * TT * 19 * 32;
  if (tid >= total) return;
  int co = (int)(tid & 31);
  int j = (int)((tid >> 5) % 19);
  int i = (int)((tid / 608) % TT);
  int b = (int)(tid / (608L * TT));
  float acc = bi[co];
  for (int kh = 0; kh < 3; ++kh)
    for (int kw = 0; kw < 3; ++kw)
      for (int ci = 0; ci < 32; ++ci)
        acc += x[((size_t)b * 1599 + 2 * i + kh) * 1248 + (2 * j + kw) * 32 + ci] *
               w[((kh * 3 + kw) * 32 + ci) * 32 + co];
  o[((size_t)b * TT + i) * DD0 + j * 32 + co] = (_Float16)acc;
}

// (K,N) f32 -> (N,K) f16 transpose for WMMA B-operands
__global__ void transpose_f16_kernel(const float* __restrict__ s,
                                     _Float16* __restrict__ d, int K, int N) {
  long tid = (long)blockIdx.x * 256 + threadIdx.x;
  if (tid >= (long)K * N) return;
  long k = tid / N, n = tid % N;
  d[n * (long)K + k] = (_Float16)s[tid];
}

// ---------------------------------------------------------------------------
// WMMA GEMM: C(MxN) = A(f16, MxK row-major) * Bt(f16, NxK)^T + bias
//   mode 0: store f32 (+bias)                     [z_in, h/c projections]
//   mode 1: bias -> BN -> ReLU, f16 (+opt f32)    [layer projection]
// Block tile: 32(M) x 512(N); 8 waves, each 32x64 (8 accumulators).
// The 32-row A tile is shared by all waves and streamed into LDS with
// double-buffered GLOBAL_LOAD_ASYNC_TO_LDS_B128 stages (ASYNCcnt overlap);
// B fragments come from global, each feeding two WMMAs.
// ---------------------------------------------------------------------------
__global__ __launch_bounds__(256) void wgemm_kernel(
    const _Float16* __restrict__ A, int lda, const _Float16* __restrict__ Bt,
    const float* __restrict__ bias, const float* __restrict__ gamma,
    const float* __restrict__ beta, const float* __restrict__ mean,
    const float* __restrict__ var, float* __restrict__ outF32,
    _Float16* __restrict__ outF16, int ldo, int M, int N, int K, int mode) {
  __shared__ _Float16 aT[2][32 * KSTAGE];  // 2 x 8 KB
  const int tid = threadIdx.x;
  const int wv = tid >> 5;
  const int lane = tid & 31;
  const int nBlk = N >> 9;  // N / 512
  const int mb = blockIdx.x / nBlk;
  const int nb = blockIdx.x % nBlk;
  const int m0 = mb << 5;
  const int n0 = (nb << 9) + (wv << 6);
  const int S = (K + KSTAGE - 1) / KSTAGE;

  auto issue = [&](int s) {
    int k0 = s * KSTAGE;
    int ck = K - k0;
    if (ck > KSTAGE) ck = KSTAGE;
    int cw = ck >> 3;        // 16B chunks per row
    int chunks = 32 * cw;
    _Float16* dst = aT[s & 1];
#pragma unroll
    for (int it = 0; it < (32 * KSTAGE / 8) / 256; ++it) {  // 2 per thread
      int idx = tid + it * 256;
      if (idx >= chunks) idx = 0;  // keep per-wave async-instr count uniform
      int row = idx / cw;
      int c = (idx % cw) << 3;
      g2l_b128(A + (size_t)(m0 + row) * lda + k0 + c, dst + row * KSTAGE + c);
    }
  };

  issue(0);
  v8f acc[8];
#pragma unroll
  for (int t = 0; t < 8; ++t) acc[t] = {};

  for (int s = 0; s < S; ++s) {
    if (s + 1 < S) {
      issue(s + 1);     // stream next stage while we compute this one
      wait_async<2>();  // oldest stage (2 instrs/wave) complete
    } else {
      wait_async<0>();
    }
    __syncthreads();
    const _Float16* at = aT[s & 1];
    const int k0 = s * KSTAGE;
    int ck = K - k0;
    if (ck > KSTAGE) ck = KSTAGE;
    for (int kk = 0; kk < ck; kk += 32) {
      v16h a0 = load_frag(at, KSTAGE, 0, kk, lane);
      v16h a1 = load_frag(at, KSTAGE, 16, kk, lane);
      __builtin_prefetch(Bt + (size_t)(n0 + (lane & 15)) * K + k0 + kk + 64, 0, 1);
#pragma unroll
      for (int t = 0; t < 4; ++t) {
        v16h b = load_frag(Bt, K, n0 + (t << 4), k0 + kk, lane);
        acc[t] = wmma_f16(a0, b, acc[t]);
        acc[4 + t] = wmma_f16(a1, b, acc[4 + t]);
      }
    }
    __syncthreads();  // stage consumed before it is overwritten
  }

  const int n = lane & 15, half = lane >> 4;
#pragma unroll
  for (int mt = 0; mt < 2; ++mt) {
#pragma unroll
    for (int t = 0; t < 4; ++t) {
      const int col = n0 + (t << 4) + n;
      float bz = bias ? bias[col] : 0.f;
      float sc = 1.f, sh = 0.f;
      if (mode == 1) {
        float inv = rsqrtf(var[col] + 1e-3f);
        sc = gamma[col] * inv;
        sh = beta[col] - mean[col] * sc;
      }
#pragma unroll
      for (int r = 0; r < 8; ++r) {
        const int row = m0 + (mt << 4) + half * 8 + r;
        float v = acc[mt * 4 + t][r] + bz;
        if (mode == 1) v = fmaxf(v * sc + sh, 0.f);
        if (outF32) outF32[(size_t)row * ldo + col] = v;
        if (outF16) outF16[(size_t)row * ldo + col] = (_Float16)v;
      }
    }
  }
}

// ---------------------------------------------------------------------------
// Grid-wide barrier over the 16 workgroups of one direction
// ---------------------------------------------------------------------------
__device__ __forceinline__ void grid_barrier(unsigned* cnt, unsigned* gen) {
  __threadfence();
  __syncthreads();
  if (threadIdx.x == 0) {
    unsigned g = *(volatile unsigned*)gen;
    if (atomicAdd(cnt, 1u) == 15u) {
      *(volatile unsigned*)cnt = 0u;  // safe: others re-arrive only after gen++
      __threadfence();
      atomicAdd(gen, 1u);
    } else {
      while (*(volatile unsigned*)gen == g) { __builtin_amdgcn_s_sleep(8); }
    }
  }
  __syncthreads();
  __threadfence();
}

// ---------------------------------------------------------------------------
// Persistent bidirectional LSTM scan for one layer.
// Grid = 32 WGs (16 fwd + 16 bwd), 128 threads (4 waves) each.
// Recurrent weight slice (128x512 f16) is async-staged into LDS once and
// stays resident for all T steps; the 32x512 hidden state is exchanged via
// ping-pong global f16 buffers and async-restaged into LDS every step.
// Wave w: batch tile = w>>1, unit half = w&1, holding the FOUR gate
// accumulators (i,f,g,o) for its units -> gate math entirely in registers;
// c and prev_out live in VGPRs across the whole scan.
// ---------------------------------------------------------------------------
__global__ __launch_bounds__(128) void lstm_scan_kernel(
    const float* __restrict__ zF, const float* __restrict__ zB,
    const _Float16* __restrict__ rkTF, const _Float16* __restrict__ rkTB,
    const float* __restrict__ maskW, const float* __restrict__ stIn,
    float* __restrict__ stOut, _Float16* __restrict__ y16,
    _Float16* __restrict__ hsh, unsigned* __restrict__ bar, int T) {
  extern __shared__ char smem[];
  _Float16* ldsB = (_Float16*)smem;                    // 128 x 512 (128 KB)
  _Float16* ldsH = (_Float16*)(smem + 128 * 512 * 2);  // 32 x 512  (32 KB)

  const int wg = blockIdx.x;
  const int dir = wg >> 4;
  const int u0 = (wg & 15) * 32;
  const int tid = threadIdx.x;
  const int wv = tid >> 5;
  const int lane = tid & 31;
  const int mtile = wv >> 1;
  const int uh = wv & 1;
  const int n = lane & 15;
  const int half = lane >> 4;
  const int bbase = mtile * 16 + half * 8;  // + r
  const int unit = u0 + uh * 16 + n;        // global unit id

  const float* z = dir ? zB : zF;
  const _Float16* rkT = dir ? rkTB : rkTF;
  _Float16* hq = hsh + (size_t)dir * 2 * (NB * UU);  // 2 ping-pong buffers
  unsigned* cnt = bar + dir * 2;
  unsigned* gen = bar + dir * 2 + 1;

  // Stage recurrent weights (async): LDS row lr = gate*32 + localUnit maps to
  // Bt row gate*512 + u0 + localUnit.
  for (int idx = tid; idx < 128 * 64; idx += 128) {  // 8-half chunks
    int lr = idx >> 6;
    int kc = (idx & 63) << 3;
    int gr = ((lr >> 5) << 9) + u0 + (lr & 31);
    g2l_b128(rkT + (size_t)gr * 512 + kc, ldsB + lr * 512 + kc);
  }
  wait_async<0>();

  const float* hin = stIn + (size_t)dir * 2 * (NB * UU);
  const float* cin = hin + NB * UU;
  float hreg[8], creg[8], poreg[8];
#pragma unroll
  for (int r = 0; r < 8; ++r) {
    int b = bbase + r;
    hreg[r] = hin[b * UU + unit];
    creg[r] = cin[b * UU + unit];
    poreg[r] = 0.f;
    hq[b * UU + unit] = (_Float16)hreg[r];  // buffer 0 for step 0
  }
  __syncthreads();

  for (int s = 0; s < T; ++s) {
    grid_barrier(cnt, gen);  // all WGs published h for this step
    const int rb = s & 1, wb = rb ^ 1;
    for (int idx = tid; idx < (NB * UU) / 8; idx += 128)
      g2l_b128(hq + rb * (NB * UU) + (idx << 3), ldsH + (idx << 3));
    wait_async<0>();
    __syncthreads();

    const int t = dir ? (T - 1 - s) : s;
    v8f acc[4];
#pragma unroll
    for (int g = 0; g < 4; ++g) {
#pragma unroll
      for (int r = 0; r < 8; ++r) {
        int b = bbase + r;
        acc[g][r] = z[((size_t)b * T + t) * GG + (g << 9) + unit];
      }
    }
    for (int k0 = 0; k0 < UU; k0 += 32) {
      v16h a = load_frag(ldsH, UU, mtile * 16, k0, lane);
#pragma unroll
      for (int g = 0; g < 4; ++g) {
        v16h bfr = load_frag(ldsB, UU, (g << 5) + uh * 16, k0, lane);
        acc[g] = wmma_f16(a, bfr, acc[g]);
      }
    }
#pragma unroll
    for (int r = 0; r < 8; ++r) {
      int b = bbase + r;
      float ig = 1.f / (1.f + __expf(-acc[0][r]));
      float fg = 1.f / (1.f + __expf(-acc[1][r]));
      float gg = tanhf(acc[2][r]);
      float og = 1.f / (1.f + __expf(-acc[3][r]));
      float cn = fg * creg[r] + ig * gg;
      float hn = og * tanhf(cn);
      bool mk = maskW[(size_t)b * T + t] != 0.f;
      creg[r] = mk ? cn : creg[r];
      hreg[r] = mk ? hn : hreg[r];
      poreg[r] = mk ? hn : poreg[r];
      y16[((size_t)b * T + t) * DY + (dir << 9) + unit] = (_Float16)poreg[r];
      hq[wb * (NB * UU) + b * UU + unit] = (_Float16)hreg[r];
    }
  }

  float* hout = stOut + (size_t)dir * 2 * (NB * UU);
  float* cout = hout + NB * UU;
#pragma unroll
  for (int r = 0; r < 8; ++r) {
    int b = bbase + r;
    hout[b * UU + unit] = hreg[r];
    cout[b * UU + unit] = creg[r];
  }
}

// ---------------------------------------------------------------------------
// Concat final states [fh|bh], [fc|bc] -> f16 for the output projections
// ---------------------------------------------------------------------------
__global__ void concat_states_kernel(const float* __restrict__ st,
                                     _Float16* __restrict__ hcat,
                                     _Float16* __restrict__ ccat) {
  int i = blockIdx.x * blockDim.x + threadIdx.x;
  if (i >= NB * DY) return;
  int b = i >> 10, u = i & 1023;
  size_t hoff = (u < UU) ? ((size_t)b * UU + u)
                         : (2 * (size_t)NB * UU + b * UU + (u - UU));
  size_t coff = (u < UU) ? ((size_t)NB * UU + b * UU + u)
                         : (3 * (size_t)NB * UU + b * UU + (u - UU));
  hcat[i] = (_Float16)st[hoff];
  ccat[i] = (_Float16)st[coff];
}

// ---------------------------------------------------------------------------
extern "C" void kernel_launch(void* const* d_in, const int* in_sizes, int n_in,
                              void* d_out, int out_size, void* d_ws,
                              size_t ws_size, hipStream_t stream) {
  (void)in_sizes; (void)n_in; (void)out_size; (void)ws_size;
  const float* audio = (const float*)d_in[0];
  const float* c1w = (const float*)d_in[1];
  const float* c1b = (const float*)d_in[2];
  const float* c2w = (const float*)d_in[3];
  const float* c2b = (const float*)d_in[4];
  const float* fk0 = (const float*)d_in[5];
  const float* bk0 = (const float*)d_in[6];
  const float* fk  = (const float*)d_in[7];   // (2,1024,2048)
  const float* bk  = (const float*)d_in[8];
  const float* fr  = (const float*)d_in[9];   // (3,512,2048)
  const float* br  = (const float*)d_in[10];
  const float* fbias = (const float*)d_in[11];
  const float* bbias = (const float*)d_in[12];
  const float* pw  = (const float*)d_in[13];  // (3,1024,1024)
  const float* pb  = (const float*)d_in[14];
  const float* gam = (const float*)d_in[15];
  const float* bet = (const float*)d_in[16];
  const float* mea = (const float*)d_in[17];
  const float* va  = (const float*)d_in[18];
  const float* hpw = (const float*)d_in[19];  // (1024,512)
  const float* hpb = (const float*)d_in[20];
  const float* cpw = (const float*)d_in[21];
  const float* cpb = (const float*)d_in[22];

  float* outX = (float*)d_out;               // (32,799,1024)
  float* outMask = outX + (size_t)MMR * DY;  // (32,799)
  float* outH = outMask + MMR;               // (32,512)
  float* outC = outH + NB * UU;              // (32,512)

  size_t off = 0;
  auto alloc = [&](size_t bytes) -> void* {
    void* p = (char*)d_ws + off;
    off += (bytes + 255) & ~(size_t)255;
    return p;
  };
  unsigned* bar = (unsigned*)alloc(64);
  float* maskW = (float*)alloc(sizeof(float) * MMR);
  float* c1 = (float*)alloc(sizeof(float) * (size_t)NB * 1599 * 39 * 32);
  _Float16* IN16 = (_Float16*)alloc(2 * (size_t)MMR * DY);
  _Float16* Y16 = (_Float16*)alloc(2 * (size_t)MMR * DY);
  float* zf = (float*)alloc(sizeof(float) * (size_t)MMR * GG);
  float* zb = (float*)alloc(sizeof(float) * (size_t)MMR * GG);
  _Float16* fkT0 = (_Float16*)alloc(2 * (size_t)GG * DD0);
  _Float16* bkT0 = (_Float16*)alloc(2 * (size_t)GG * DD0);
  _Float16* fkT1 = (_Float16*)alloc(2 * (size_t)GG * DY);
  _Float16* fkT2 = (_Float16*)alloc(2 * (size_t)GG * DY);
  _Float16* bkT1 = (_Float16*)alloc(2 * (size_t)GG * DY);
  _Float16* bkT2 = (_Float16*)alloc(2 * (size_t)GG * DY);
  _Float16* frT[3]; _Float16* brT[3]; _Float16* pT[3];
  for (int l = 0; l < 3; ++l) frT[l] = (_Float16*)alloc(2 * (size_t)GG * UU);
  for (int l = 0; l < 3; ++l) brT[l] = (_Float16*)alloc(2 * (size_t)GG * UU);
  for (int l = 0; l < 3; ++l) pT[l] = (_Float16*)alloc(2 * (size_t)DY * DY);
  _Float16* hpT = (_Float16*)alloc(2 * (size_t)UU * DY);
  _Float16* cpT = (_Float16*)alloc(2 * (size_t)UU * DY);
  _Float16* hsh = (_Float16*)alloc(2 * (size_t)2 * 2 * NB * UU);
  float* stA = (float*)alloc(sizeof(float) * 4 * NB * UU);
  float* stB = (float*)alloc(sizeof(float) * 4 * NB * UU);
  _Float16* hcat = (_Float16*)alloc(2 * (size_t)NB * DY);
  _Float16* ccat = (_Float16*)alloc(2 * (size_t)NB * DY);

  (void)hipMemsetAsync(bar, 0, 64, stream);
  (void)hipMemsetAsync(stA, 0, sizeof(float) * 4 * NB * UU, stream);

  mask_kernel<<<(MMR + 255) / 256, 256, 0, stream>>>(audio, maskW, outMask);
  {
    long ntot = (long)NB * 1599 * 39 * 32;
    conv1_kernel<<<(unsigned)((ntot + 255) / 256), 256, 0, stream>>>(audio, c1w,
                                                                     c1b, c1);
  }
  {
    long ntot = (long)NB * TT * 19 * 32;
    conv2_kernel<<<(unsigned)((ntot + 255) / 256), 256, 0, stream>>>(c1, c2w,
                                                                     c2b, IN16);
  }

  auto tr = [&](const float* s, _Float16* d, int K, int N) {
    long ntot = (long)K * N;
    transpose_f16_kernel<<<(unsigned)((ntot + 255) / 256), 256, 0, stream>>>(
        s, d, K, N);
  };
  tr(fk0, fkT0, DD0, GG); tr(bk0, bkT0, DD0, GG);
  tr(fk, fkT1, DY, GG);   tr(fk + (size_t)DY * GG, fkT2, DY, GG);
  tr(bk, bkT1, DY, GG);   tr(bk + (size_t)DY * GG, bkT2, DY, GG);
  for (int l = 0; l < 3; ++l) tr(fr + (size_t)l * UU * GG, frT[l], UU, GG);
  for (int l = 0; l < 3; ++l) tr(br + (size_t)l * UU * GG, brT[l], UU, GG);
  for (int l = 0; l < 3; ++l) tr(pw + (size_t)l * DY * DY, pT[l], DY, DY);
  tr(hpw, hpT, DY, UU); tr(cpw, cpT, DY, UU);

  auto gemm = [&](const _Float16* A, int lda, const _Float16* Bt,
                  const float* bias, const float* g_, const float* b_,
                  const float* m_, const float* v_, float* o32, _Float16* o16,
                  int ldo, int M, int N, int K, int mode) {
    unsigned blocks = (unsigned)((long)(M / 32) * (N / 512));
    wgemm_kernel<<<blocks, 256, 0, stream>>>(A, lda, Bt, bias, g_, b_, m_, v_,
                                             o32, o16, ldo, M, N, K, mode);
  };

  const _Float16* fkTl[3] = {fkT0, fkT1, fkT2};
  const _Float16* bkTl[3] = {bkT0, bkT1, bkT2};
  float* stIn = stA;
  float* stOut = stB;
  const size_t scanLds = (size_t)(128 * 512 + 32 * 512) * 2;  // 160 KB

  for (int l = 0; l < 3; ++l) {
    int Kl = (l == 0) ? DD0 : DY;
    gemm(IN16, Kl, fkTl[l], fbias + (size_t)l * GG, nullptr, nullptr, nullptr,
         nullptr, zf, nullptr, GG, MMR, GG, Kl, 0);
    gemm(IN16, Kl, bkTl[l], bbias + (size_t)l * GG, nullptr, nullptr, nullptr,
         nullptr, zb, nullptr, GG, MMR, GG, Kl, 0);
    lstm_scan_kernel<<<32, 128, scanLds, stream>>>(zf, zb, frT[l], brT[l],
                                                   maskW, stIn, stOut, Y16, hsh,
                                                   bar, TT);
    gemm(Y16, DY, pT[l], pb + (size_t)l * DY, gam + (size_t)l * DY,
         bet + (size_t)l * DY, mea + (size_t)l * DY, va + (size_t)l * DY,
         (l == 2) ? outX : nullptr, IN16, DY, MMR, DY, DY, 1);
    float* t = stIn; stIn = stOut; stOut = t;  // final states land in stIn
  }

  concat_states_kernel<<<(NB * DY + 255) / 256, 256, 0, stream>>>(stIn, hcat,
                                                                  ccat);
  gemm(hcat, DY, hpT, hpb, nullptr, nullptr, nullptr, nullptr, outH, nullptr,
       UU, NB, UU, DY, 0);
  gemm(ccat, DY, cpT, cpb, nullptr, nullptr, nullptr, nullptr, outC, nullptr,
       UU, NB, UU, DY, 0);
}
